// RWKV_Tmix_x070_58454504899042
// MI455X (gfx1250) — matrix operand order repaired and checked by reading, not compiled, not run on hardware
//
#include <hip/hip_runtime.h>
#include <hip/hip_bf16.h>
#include <math.h>

typedef __attribute__((ext_vector_type(16))) __bf16 v16bf;
typedef __attribute__((ext_vector_type(8)))  __bf16 v8bf;
typedef __attribute__((ext_vector_type(8)))  float  v8f;

#define Bdim  8
#define Tdim  2048
#define Cdim  1024
#define Hn    16
#define HN    64
#define Mrows (Bdim * Tdim)

#define EP_NONE     0
#define EP_TANH     1
#define EP_SIG_BIAS 2
#define EP_SIG      3
#define EP_DECAY    4

#define MODE_MIX   0
#define MODE_PLAIN 1

#define LDS_STR 40   // 32 + 8 halves pad; keeps 16B alignment for v8bf loads

__device__ __forceinline__ float apply_ep(float v, const float* bias, int n, int ep) {
  if (ep == EP_TANH) return tanhf(v);
  if (ep == EP_SIG_BIAS) { float z = bias[n] + v; return 1.f / (1.f + expf(-z)); }
  if (ep == EP_SIG) { return 1.f / (1.f + expf(-v)); }
  if (ep == EP_DECAY) {
    float d  = bias[n] + v;               // time_decay + mlp
    float sp = log1pf(expf(-d));          // softplus(-d)
    float w  = -sp - 0.5f;                // reference w
    return expf(-expf(w));                // decay factor used by recurrence
  }
  return v;
}

// ---------------------------------------------------------------------------
// fp32 -> bf16 (bit pattern) weight pre-conversion (one pass, ~10 MB total)
// ---------------------------------------------------------------------------
__global__ __launch_bounds__(256)
void f32_to_bf16(const float* __restrict__ s, unsigned short* __restrict__ d, int n) {
  int i = blockIdx.x * blockDim.x + threadIdx.x;
  if (i < n) {
    __bf16 b = (__bf16)s[i];
    d[i] = __builtin_bit_cast(unsigned short, b);
  }
}

// ---------------------------------------------------------------------------
// Generic bf16 WMMA GEMM: D[M,N] = A[M,K] @ Wbf[K,N] (+ fused epilogue).
// A is fp32 (converted in-flight); B is pre-converted bf16 (halves traffic).
// mode MIX: A row m is x[m] + (x[m-1]-x[m])*maa (time-shift mixing, K==C).
// Block: 128 thr (4 waves), 64x64 tile; wave: 32x32 via 2x2 16x16x32 WMMAs.
// ---------------------------------------------------------------------------
__global__ __launch_bounds__(128)
void gemm_bf16_wmma(const float* __restrict__ Abuf, const float* __restrict__ X,
                    const float* __restrict__ maa,  const unsigned short* __restrict__ Wbf,
                    const float* __restrict__ bias, float* __restrict__ D,
                    int Md, int Nd, int Kd, int mode, int ep)
{
  __shared__ __align__(16) __bf16         ldsA[64 * LDS_STR];   // [m][k]
  __shared__ __align__(16) unsigned short ldsB[64 * LDS_STR];   // [n][k] (transposed)

  const int tid   = threadIdx.x;
  const int lane  = tid & 31;
  const int wid   = tid >> 5;
  const int waveM = wid >> 1, waveN = wid & 1;
  const int bm = blockIdx.y * 64;
  const int bn = blockIdx.x * 64;

  v8f acc[2][2];
  #pragma unroll
  for (int i = 0; i < 2; i++)
    #pragma unroll
    for (int j = 0; j < 2; j++)
      #pragma unroll
      for (int e = 0; e < 8; e++) acc[i][j][e] = 0.f;

  const int arow  = tid >> 1;         // 0..63  (A tile row)
  const int ahalf = (tid & 1) * 16;   // 0/16   (A tile k half)
  const int bk    = tid >> 2;         // 0..31  (B tile k row)
  const int bseg  = (tid & 3) * 16;   // 0..48  (B tile n segment)

  for (int k0 = 0; k0 < Kd; k0 += 32) {
    // ---- cooperative A tile load (64 x 32), fp32 -> bf16, fused mix ----
    {
      const int m = bm + arow;
      const int kc = k0 + ahalf;
      if (mode == MODE_MIX) {
        const float* xr = X + (size_t)m * Kd;
        const bool hp = (m % Tdim) != 0;
        #pragma unroll
        for (int j = 0; j < 16; j += 4) {
          float4 xv = *(const float4*)(xr + kc + j);
          float4 pv = make_float4(0.f, 0.f, 0.f, 0.f);
          if (hp) pv = *(const float4*)(xr - Kd + kc + j);
          float4 mv = *(const float4*)(maa + kc + j);
          __bf16* dst = &ldsA[arow * LDS_STR + ahalf + j];
          dst[0] = (__bf16)(xv.x + (pv.x - xv.x) * mv.x);
          dst[1] = (__bf16)(xv.y + (pv.y - xv.y) * mv.y);
          dst[2] = (__bf16)(xv.z + (pv.z - xv.z) * mv.z);
          dst[3] = (__bf16)(xv.w + (pv.w - xv.w) * mv.w);
        }
      } else {
        const float* ar = Abuf + (size_t)m * Kd + kc;
        #pragma unroll
        for (int j = 0; j < 16; j += 4) {
          float4 av = *(const float4*)(ar + j);
          __bf16* dst = &ldsA[arow * LDS_STR + ahalf + j];
          dst[0] = (__bf16)av.x; dst[1] = (__bf16)av.y;
          dst[2] = (__bf16)av.z; dst[3] = (__bf16)av.w;
        }
      }
    }
    // ---- cooperative B tile load (32 x 64, bf16) -> transposed ldsB[n][k] ----
    {
      const int kg = k0 + bk;
      const int nc = bn + bseg;
      if (nc + 15 < Nd) {
        const uint4* wr = (const uint4*)(Wbf + (size_t)kg * Nd + nc);
        uint4 wa = wr[0];   // bf16 pairs: n = nc+0..7
        uint4 wb = wr[1];   // n = nc+8..15
        unsigned u[8] = {wa.x, wa.y, wa.z, wa.w, wb.x, wb.y, wb.z, wb.w};
        #pragma unroll
        for (int j = 0; j < 8; j++) {
          ldsB[(bseg + 2 * j + 0) * LDS_STR + bk] = (unsigned short)(u[j] & 0xffffu);
          ldsB[(bseg + 2 * j + 1) * LDS_STR + bk] = (unsigned short)(u[j] >> 16);
        }
      } else {
        #pragma unroll
        for (int j = 0; j < 16; j++)
          ldsB[(bseg + j) * LDS_STR + bk] = 0;
      }
    }
    __syncthreads();

    // ---- build fragments per ISA layout and issue 4 WMMAs ----
    const int l15   = lane & 15;
    const int khalf = (lane >> 4) * 8;    // A: lanes 16-31 hold K+8 groups
    const int kb16  = (lane >> 4) * 16;   // unused; B is read with khalf, same k order as A
    v16bf af[2], bf[2];
    #pragma unroll
    for (int f = 0; f < 2; f++) {
      const int mrow = waveM * 32 + f * 16 + l15;
      v8bf alo = *(const v8bf*)&ldsA[mrow * LDS_STR + khalf];        // K=khalf..+7
      v8bf ahi = *(const v8bf*)&ldsA[mrow * LDS_STR + 16 + khalf];   // K=16+khalf..
      #pragma unroll
      for (int e = 0; e < 8; e++) { af[f][e] = alo[e]; af[f][e + 8] = ahi[e]; }
      const int nrow = waveN * 32 + f * 16 + l15;
      v8bf blo = *(const v8bf*)&ldsB[nrow * LDS_STR + khalf];        // K=khalf..+7
      v8bf bhi = *(const v8bf*)&ldsB[nrow * LDS_STR + 16 + khalf];   // K=16+khalf..
      #pragma unroll
      for (int e = 0; e < 8; e++) { bf[f][e] = blo[e]; bf[f][e + 8] = bhi[e]; }
    }
    #pragma unroll
    for (int i = 0; i < 2; i++)
      #pragma unroll
      for (int j = 0; j < 2; j++)
        acc[i][j] = __builtin_amdgcn_wmma_f32_16x16x32_bf16(
            false, af[i], false, bf[j], (short)0, acc[i][j], false, false);
    __syncthreads();
  }

  // ---- store with fused epilogue (C/D layout: m = e + 8*(lane>>4), n = lane&15)
  const int l15 = lane & 15;
  const int mo  = (lane >> 4) * 8;
  #pragma unroll
  for (int i = 0; i < 2; i++)
    #pragma unroll
    for (int j = 0; j < 2; j++) {
      const int cm = bm + waveM * 32 + i * 16 + mo;
      const int cn = bn + waveN * 32 + j * 16 + l15;
      if (cn < Nd) {
        #pragma unroll
        for (int e = 0; e < 8; e++)
          D[(size_t)(cm + e) * Nd + cn] = apply_ep(acc[i][j][e], bias, cn, ep);
      }
    }
}

// ---------------------------------------------------------------------------
// Per-(b,t) row prep: kk = normalize_head(k*misc_kkk); k <- k*(1+(a-1)*misc_a);
// v <- v + (v0 - v)*sv.  One wave32 per head (2 elems/lane), shuffle reduce.
// ---------------------------------------------------------------------------
__global__ __launch_bounds__(512)
void prep_kernel(float* __restrict__ Kb, const float* __restrict__ Ab,
                 float* __restrict__ Vb, const float* __restrict__ SVb,
                 const float* __restrict__ Vres, float* __restrict__ KKb,
                 const float* __restrict__ misc_kkk, const float* __restrict__ misc_a)
{
  const int row = blockIdx.x;
  const int wid = threadIdx.x >> 5, lane = threadIdx.x & 31;
  const int c0 = wid * HN + lane, c1 = c0 + 32;
  const size_t i0 = (size_t)row * Cdim + c0, i1 = i0 + 32;

  float k0 = Kb[i0], k1 = Kb[i1];
  float kk0 = k0 * misc_kkk[c0], kk1 = k1 * misc_kkk[c1];
  float ss = kk0 * kk0 + kk1 * kk1;
  #pragma unroll
  for (int m = 16; m >= 1; m >>= 1) ss += __shfl_xor(ss, m, 32);
  float inv = 1.f / fmaxf(sqrtf(ss), 1e-12f);
  KKb[i0] = kk0 * inv; KKb[i1] = kk1 * inv;

  float a0 = Ab[i0], a1 = Ab[i1];
  Kb[i0] = k0 * (1.f + (a0 - 1.f) * misc_a[c0]);
  Kb[i1] = k1 * (1.f + (a1 - 1.f) * misc_a[c1]);

  float vv0 = Vb[i0], vv1 = Vb[i1];
  Vb[i0] = vv0 + (Vres[i0] - vv0) * SVb[i0];
  Vb[i1] = vv1 + (Vres[i1] - vv1) * SVb[i1];
}

// ---------------------------------------------------------------------------
// RWKV7 recurrence: one block per (b,h); thread i owns state row i (64 f32
// in VGPRs). Per-step coefficient vectors (r,w,k,kk,a,v = 6x64 f32) are
// double-buffered into LDS with CDNA5 GLOBAL_LOAD_ASYNC_TO_LDS_B64 prefetch:
// while step t computes from buf[p], step t+1 streams into buf[p^1]; the
// counted wait (s_wait_asynccnt 3) only drains step t's three async ops,
// keeping the next prefetch in flight across the serial dependency chain.
// ---------------------------------------------------------------------------
__global__ __launch_bounds__(64)
void rwkv7_recurrence(const float* __restrict__ R,  const float* __restrict__ Wf,
                      const float* __restrict__ Ka, const float* __restrict__ KK,
                      const float* __restrict__ Aa, const float* __restrict__ Vv,
                      float* __restrict__ Y)
{
  const int bh = blockIdx.x;            // 0..B*H-1
  const int b = bh / Hn, h = bh % Hn;
  const int i = threadIdx.x;            // state row
  const int wid = i >> 5, lane = i & 31;

  __shared__ __align__(16) float shbuf[2][6 * HN];   // double buffer, 3 KB

  // --- per-lane async source/dest addresses: 2 waves x 3 ops x 32 lanes x 8B
  //     flat element f covers [arr 0..5][elem 0..63], 2 floats per lane.
  unsigned long long ga[3];
  unsigned la[3];
  const size_t base0 = (size_t)(b * Tdim) * Cdim + h * HN;
  #pragma unroll
  for (int o = 0; o < 3; o++) {
    int f   = ((wid * 3 + o) * 32 + lane) * 2;
    int arr = f >> 6, el = f & 63;
    const float* bp = R;
    if (arr == 1) bp = Wf;
    else if (arr == 2) bp = Ka;
    else if (arr == 3) bp = KK;
    else if (arr == 4) bp = Aa;
    else if (arr == 5) bp = Vv;
    ga[o] = (unsigned long long)(bp + base0 + el);
    la[o] = (unsigned)(uintptr_t)&shbuf[0][f];
  }

  float st[HN];
  #pragma unroll
  for (int j = 0; j < HN; j++) st[j] = 0.f;

  // prologue: stream step 0 into buf0
  #pragma unroll
  for (int o = 0; o < 3; o++) {
    asm volatile("global_load_async_to_lds_b64 %0, %1, off"
                 :: "v"(la[o]), "v"(ga[o]) : "memory");
    ga[o] += (unsigned long long)(Cdim * sizeof(float));
  }

  int p = 0;
  for (int t = 0; t < Tdim; t++) {
    if (t + 1 < Tdim) {
      const unsigned boff = (unsigned)((p ^ 1) * 6 * HN * sizeof(float));
      #pragma unroll
      for (int o = 0; o < 3; o++) {
        asm volatile("global_load_async_to_lds_b64 %0, %1, off"
                     :: "v"(la[o] + boff), "v"(ga[o]) : "memory");
        ga[o] += (unsigned long long)(Cdim * sizeof(float));
      }
      asm volatile("s_wait_asynccnt 0x3" ::: "memory");   // step t landed
    } else {
      asm volatile("s_wait_asynccnt 0x0" ::: "memory");
    }
    __syncthreads();

    const float* sh   = &shbuf[p][0];
    const float* shr  = sh + 0 * HN;
    const float* shw  = sh + 1 * HN;
    const float* shk  = sh + 2 * HN;
    const float* shkk = sh + 3 * HN;
    const float* sha  = sh + 4 * HN;
    const float  vi   = sh[5 * HN + i];

    float sa = 0.f;
    #pragma unroll
    for (int j = 0; j < HN; j++) sa -= st[j] * shkk[j];   // state @ (-kk)

    float out = 0.f;
    #pragma unroll
    for (int j = 0; j < HN; j++) {
      st[j] = st[j] * shw[j] + sa * (shkk[j] * sha[j]) + vi * shk[j];
      out += st[j] * shr[j];
    }
    const size_t base = base0 + (size_t)t * Cdim;
    Y[base + i] = out;
    __syncthreads();    // all readers done before buf[p] is refilled next iter
    p ^= 1;
  }
}

// ---------------------------------------------------------------------------
// Per-head GroupNorm + rkv residual + gate multiply  ->  P = (gn(y)+rkv)*g
// ---------------------------------------------------------------------------
__global__ __launch_bounds__(512)
void gn_rkv_gate(const float* __restrict__ Y,  const float* __restrict__ R,
                 const float* __restrict__ Ka, const float* __restrict__ Vv,
                 const float* __restrict__ G,  const float* __restrict__ faaaa,
                 const float* __restrict__ lnw, const float* __restrict__ lnb,
                 float* __restrict__ P)
{
  const int row = blockIdx.x;
  const int wid = threadIdx.x >> 5, lane = threadIdx.x & 31;
  const int c0 = wid * HN + lane, c1 = c0 + 32;
  const size_t i0 = (size_t)row * Cdim + c0, i1 = i0 + 32;

  float y0 = Y[i0], y1 = Y[i1];
  float s1 = y0 + y1, s2 = y0 * y0 + y1 * y1;
  #pragma unroll
  for (int m = 16; m >= 1; m >>= 1) {
    s1 += __shfl_xor(s1, m, 32);
    s2 += __shfl_xor(s2, m, 32);
  }
  const float mu  = s1 * (1.f / HN);
  const float var = s2 * (1.f / HN) - mu * mu;
  const float rs  = rsqrtf(var + 6.4e-4f);   // 1e-5 * 8^2
  float yn0 = (y0 - mu) * rs * lnw[c0] + lnb[c0];
  float yn1 = (y1 - mu) * rs * lnw[c1] + lnb[c1];

  float rk = R[i0] * Ka[i0] * faaaa[c0] + R[i1] * Ka[i1] * faaaa[c1];
  #pragma unroll
  for (int m = 16; m >= 1; m >>= 1) rk += __shfl_xor(rk, m, 32);

  P[i0] = (yn0 + rk * Vv[i0]) * G[i0];
  P[i1] = (yn1 + rk * Vv[i1]) * G[i1];
}

// ---------------------------------------------------------------------------
extern "C" void kernel_launch(void* const* d_in, const int* in_sizes, int n_in,
                              void* d_out, int out_size, void* d_ws, size_t ws_size,
                              hipStream_t stream)
{
  const float* x          = (const float*)d_in[0];
  const float* v0         = (const float*)d_in[1];
  const float* maa_r      = (const float*)d_in[2];
  const float* maa_w      = (const float*)d_in[3];
  const float* maa_k      = (const float*)d_in[4];
  const float* maa_v      = (const float*)d_in[5];
  const float* maa_a      = (const float*)d_in[6];
  const float* maa_g      = (const float*)d_in[7];
  const float* time_decay = (const float*)d_in[8];
  const float* faaaa      = (const float*)d_in[9];
  const float* time_aaaaa = (const float*)d_in[10];
  const float* decay_w1   = (const float*)d_in[11];
  const float* decay_w2   = (const float*)d_in[12];
  const float* aaa_w1     = (const float*)d_in[13];
  const float* aaa_w2     = (const float*)d_in[14];
  const float* gate_w1    = (const float*)d_in[15];
  const float* gate_w2    = (const float*)d_in[16];
  const float* mv_w1      = (const float*)d_in[17];
  const float* mv_w2      = (const float*)d_in[18];
  const float* misc_v     = (const float*)d_in[19];
  const float* misc_kkk   = (const float*)d_in[20];
  const float* misc_a     = (const float*)d_in[21];
  const float* W_r        = (const float*)d_in[22];
  const float* W_k        = (const float*)d_in[23];
  const float* W_v        = (const float*)d_in[24];
  const float* W_out      = (const float*)d_in[25];
  const float* ln_w       = (const float*)d_in[26];
  const float* ln_b       = (const float*)d_in[27];

  float* ws = (float*)d_ws;
  const size_t MC = (size_t)Mrows * Cdim;
  float* bufR  = ws + 0 * MC;
  float* bufK  = ws + 1 * MC;   // raw k, later modified k
  float* bufV  = ws + 2 * MC;   // raw v, later residual-mixed v
  float* bufA  = ws + 3 * MC;   // sigmoid "a"
  float* bufSV = ws + 4 * MC;   // mv sigmoid gate
  float* bufG  = ws + 5 * MC;   // output gate g
  float* bufW  = ws + 6 * MC;   // decay factor exp(-exp(w)); reused as P later
  float* bufKK = ws + 7 * MC;   // normalized kk
  float* bufY  = ws + 8 * MC;   // recurrence output
  float* bufT1 = ws + 9 * MC;   // [M,128] small-MLP scratch (fits below 10*MC)

  // bf16 weight area (ushort), after 10*MC floats
  unsigned short* wb = (unsigned short*)(ws + 10 * MC);
  const int CC = Cdim * Cdim;
  unsigned short* w_r   = wb;                 // CC
  unsigned short* w_k   = w_r + CC;
  unsigned short* w_v   = w_k + CC;
  unsigned short* w_out = w_v + CC;
  unsigned short* w_dw1 = w_out + CC;         // C*64
  unsigned short* w_dw2 = w_dw1 + Cdim * 64;  // 64*C
  unsigned short* w_aw1 = w_dw2 + Cdim * 64;
  unsigned short* w_aw2 = w_aw1 + Cdim * 64;
  unsigned short* w_gw1 = w_aw2 + Cdim * 64;  // C*128
  unsigned short* w_gw2 = w_gw1 + Cdim * 128; // 128*C
  unsigned short* w_mw1 = w_gw2 + Cdim * 128; // C*32
  unsigned short* w_mw2 = w_mw1 + Cdim * 32;  // 32*C

  #define CVT(src, dst, n) \
    f32_to_bf16<<<dim3(((n) + 255) / 256), dim3(256), 0, stream>>>(src, dst, n)
  CVT(W_r, w_r, CC);       CVT(W_k, w_k, CC);
  CVT(W_v, w_v, CC);       CVT(W_out, w_out, CC);
  CVT(decay_w1, w_dw1, Cdim * 64);  CVT(decay_w2, w_dw2, Cdim * 64);
  CVT(aaa_w1, w_aw1, Cdim * 64);    CVT(aaa_w2, w_aw2, Cdim * 64);
  CVT(gate_w1, w_gw1, Cdim * 128);  CVT(gate_w2, w_gw2, Cdim * 128);
  CVT(mv_w1, w_mw1, Cdim * 32);     CVT(mv_w2, w_mw2, Cdim * 32);
  #undef CVT

  const dim3 blk(128);
  #define GEMM(Ab, Xp, mp, Wp, bp, Dp, Nd, Kd, mode, ep)                        \
    gemm_bf16_wmma<<<dim3(((Nd) + 63) / 64, Mrows / 64), blk, 0, stream>>>(     \
        Ab, Xp, mp, Wp, bp, Dp, Mrows, Nd, Kd, mode, ep)

  // decay MLP: tanh(mix_w @ w1) @ w2 -> decay activation -> exp(-exp(w))
  GEMM(nullptr, x, maa_w, w_dw1, nullptr, bufT1, 64, Cdim, MODE_MIX, EP_TANH);
  GEMM(bufT1, nullptr, nullptr, w_dw2, time_decay, bufW, Cdim, 64, MODE_PLAIN, EP_DECAY);
  // a = sigmoid(time_aaaaa + mix_a @ w1 @ w2)
  GEMM(nullptr, x, maa_a, w_aw1, nullptr, bufT1, 64, Cdim, MODE_MIX, EP_NONE);
  GEMM(bufT1, nullptr, nullptr, w_aw2, time_aaaaa, bufA, Cdim, 64, MODE_PLAIN, EP_SIG_BIAS);
  // sv = sigmoid(time_misc_v + mix_v @ mv_w1 @ mv_w2)
  GEMM(nullptr, x, maa_v, w_mw1, nullptr, bufT1, 32, Cdim, MODE_MIX, EP_NONE);
  GEMM(bufT1, nullptr, nullptr, w_mw2, misc_v, bufSV, Cdim, 32, MODE_PLAIN, EP_SIG_BIAS);
  // g = sigmoid(mix_g @ gate_w1) @ gate_w2
  GEMM(nullptr, x, maa_g, w_gw1, nullptr, bufT1, 128, Cdim, MODE_MIX, EP_SIG);
  GEMM(bufT1, nullptr, nullptr, w_gw2, nullptr, bufG, Cdim, 128, MODE_PLAIN, EP_NONE);
  // big projections
  GEMM(nullptr, x, maa_r, w_r, nullptr, bufR, Cdim, Cdim, MODE_MIX, EP_NONE);
  GEMM(nullptr, x, maa_k, w_k, nullptr, bufK, Cdim, Cdim, MODE_MIX, EP_NONE);
  GEMM(nullptr, x, maa_v, w_v, nullptr, bufV, Cdim, Cdim, MODE_MIX, EP_NONE);

  prep_kernel<<<dim3(Mrows), dim3(512), 0, stream>>>(
      bufK, bufA, bufV, bufSV, v0, bufKK, misc_kkk, misc_a);

  rwkv7_recurrence<<<dim3(Bdim * Hn), dim3(64), 0, stream>>>(
      bufR, bufW, bufK, bufKK, bufA, bufV, bufY);

  gn_rkv_gate<<<dim3(Mrows), dim3(512), 0, stream>>>(
      bufY, bufR, bufK, bufV, bufG, faaaa, ln_w, ln_b, bufW /* P aliases dead W */);

  GEMM(bufW, nullptr, nullptr, w_out, nullptr, (float*)d_out, Cdim, Cdim, MODE_PLAIN, EP_NONE);

  // second tuple output: v0 passthrough
  hipMemcpyAsync((float*)d_out + MC, v0, MC * sizeof(float),
                 hipMemcpyDeviceToDevice, stream);
  #undef GEMM
}
